// LlamaAttentionTriton_38852274160172
// MI455X (gfx1250) — compile-verified, matrix operand order checked by reading, not run on hardware
//
#include <hip/hip_runtime.h>
#include <stdint.h>

// ---------- CDNA5 WMMA / TDM types ----------
typedef __attribute__((ext_vector_type(16))) __bf16 v16bf;
typedef __attribute__((ext_vector_type(8)))  float  v8f;
typedef __attribute__((ext_vector_type(4)))  unsigned int v4u;
typedef __attribute__((ext_vector_type(8)))  int    v8i;
typedef __attribute__((ext_vector_type(4)))  int    v4i;

#define NUM_HEADS    32
#define NUM_KV_HEADS 8
#define HEAD_DIM     128
#define BATCH        2
#define SEQ          2048
#define HID          4096
#define DQ           (NUM_HEADS * HEAD_DIM)     // 4096
#define DKV          (NUM_KV_HEADS * HEAD_DIM)  // 1024
#define MROWS        (BATCH * SEQ)              // 4096

// LDS row strides (ushorts) after TDM pad: K rows 256B+32B, V rows 64B+32B
#define KROW 144
#define VROW 48

__device__ __forceinline__ unsigned short f2bf(float f) {
    union { float f; uint32_t u; } c; c.f = f;
    uint32_t u = c.u;
    uint32_t r = u + 0x7FFFu + ((u >> 16) & 1u);  // round-to-nearest-even
    return (unsigned short)(r >> 16);
}

__device__ __forceinline__ v8f wmma_bf16(v16bf a, v16bf b, v8f c) {
    return __builtin_amdgcn_wmma_f32_16x16x32_bf16(
        false, a, false, b, (short)0, c, false, false);
}

// ---- TDM: 2-D bf16 tile load (dim0 contiguous, dim1 rows of stride0 elems)
// pad_i: pad_interval code (dwords = 2<<code), pad_a: pad_amount code (dwords = code+1)
__device__ __forceinline__ void tdm_load_2d(uint32_t lds_addr, const void* gptr,
                                            uint32_t dim0, uint32_t dim1,
                                            uint32_t stride0,
                                            uint32_t pad_i, uint32_t pad_a) {
    uint64_t ga = (uint64_t)(uintptr_t)gptr;
    v4u g0;
    g0.x = 1u;                                    // count=1, user load
    g0.y = lds_addr;                              // LDS byte address
    g0.z = (uint32_t)ga;                          // global_addr[31:0]
    g0.w = ((uint32_t)(ga >> 32) & 0x01FFFFFFu) | (2u << 30);  // addr[56:32], type=2
    v8i g1;
    g1[0] = (int)((1u << 16) | (1u << 20) | (pad_i << 22) | (pad_a << 25)); // 2B data, pad
    g1[1] = (int)((dim0 & 0xFFFFu) << 16);                    // tensor_dim0[15:0]
    g1[2] = (int)(((dim0 >> 16) & 0xFFFFu) | ((dim1 & 0xFFFFu) << 16));
    g1[3] = (int)(((dim1 >> 16) & 0xFFFFu) | ((dim0 & 0xFFFFu) << 16)); // tile_dim0 = dim0
    g1[4] = (int)(dim1 & 0xFFFFu);                             // tile_dim1 = dim1
    g1[5] = (int)stride0;                                      // tensor_dim0_stride lo
    g1[6] = 0;
    g1[7] = 0;
    v4i z4 = {0, 0, 0, 0};
    v8i z8 = {0, 0, 0, 0, 0, 0, 0, 0};
    __builtin_amdgcn_tensor_load_to_lds(g0, g1, z4, z4, z8, 0);
}

// ---------- 1) fp32 -> bf16 elementwise (8 elems / thread) ----------
__global__ void k_f32_to_bf16(const float* __restrict__ in,
                              unsigned short* __restrict__ out, int n) {
    int i = (blockIdx.x * blockDim.x + threadIdx.x) * 8;
    if (i >= n) return;
    const float4* p = reinterpret_cast<const float4*>(in + i);
    float4 a = p[0], b = p[1];
    uint4 o;
    o.x = (uint32_t)f2bf(a.x) | ((uint32_t)f2bf(a.y) << 16);
    o.y = (uint32_t)f2bf(a.z) | ((uint32_t)f2bf(a.w) << 16);
    o.z = (uint32_t)f2bf(b.x) | ((uint32_t)f2bf(b.y) << 16);
    o.w = (uint32_t)f2bf(b.z) | ((uint32_t)f2bf(b.w) << 16);
    *reinterpret_cast<uint4*>(out + i) = o;
}

// ---------- 2) fp32 [K][N] -> bf16 [N][K] (LDS-tiled transpose) ----------
__global__ __launch_bounds__(256) void k_transpose_w(const float* __restrict__ in,
                                                     unsigned short* __restrict__ out,
                                                     int K, int N) {
    __shared__ float tile[32][33];
    int n = blockIdx.x * 32 + threadIdx.x;
    int k = blockIdx.y * 32 + threadIdx.y;
    #pragma unroll
    for (int j = 0; j < 32; j += 8)
        tile[threadIdx.y + j][threadIdx.x] = in[(size_t)(k + j) * N + n];
    __syncthreads();
    int ko = blockIdx.y * 32 + threadIdx.x;
    int no = blockIdx.x * 32 + threadIdx.y;
    #pragma unroll
    for (int j = 0; j < 32; j += 8)
        out[(size_t)(no + j) * K + ko] = f2bf(tile[threadIdx.x][threadIdx.y + j]);
}

// ---------- 4) bf16 V [B*S][DKV] -> Vt [B][DKV][S] ----------
__global__ __launch_bounds__(256) void k_transpose_v(const unsigned short* __restrict__ v,
                                                     unsigned short* __restrict__ vt) {
    __shared__ unsigned short tile[32][33];
    int b = blockIdx.z;
    int c = blockIdx.x * 32 + threadIdx.x;
    int s = blockIdx.y * 32 + threadIdx.y;
    #pragma unroll
    for (int j = 0; j < 32; j += 8)
        tile[threadIdx.y + j][threadIdx.x] = v[(size_t)(b * SEQ + s + j) * DKV + c];
    __syncthreads();
    int so = blockIdx.y * 32 + threadIdx.x;
    int co = blockIdx.x * 32 + threadIdx.y;
    #pragma unroll
    for (int j = 0; j < 32; j += 8)
        vt[((size_t)b * DKV + co + j) * SEQ + so] = tile[threadIdx.x][threadIdx.y + j];
}

// ---------- 3/6) GEMM: C[M][N] = A[M][K] * Bt[N][K]^T, bf16 in, f32 acc ----------
template <bool BF16_OUT>
__global__ __launch_bounds__(128) void k_gemm(const unsigned short* __restrict__ A,
                                              const unsigned short* __restrict__ Bt,
                                              void* __restrict__ Cv,
                                              int Mm, int Nn, int Kk) {
    const int lane  = threadIdx.x & 31;
    const int wave  = threadIdx.x >> 5;
    const int m0    = blockIdx.y * 128 + (wave >> 1) * 64;
    const int n0    = blockIdx.x * 128 + (wave & 1) * 64;
    const int lrow  = lane & 15;
    const int khalf = (lane >> 4) * 16;

    v8f acc[4][4];
    #pragma unroll
    for (int i = 0; i < 4; ++i)
        #pragma unroll
        for (int j = 0; j < 4; ++j) acc[i][j] = v8f{};

    const unsigned short* Ap = A  + (size_t)(m0 + lrow) * Kk + khalf;
    const unsigned short* Bp = Bt + (size_t)(n0 + lrow) * Kk + khalf;

    for (int k0 = 0; k0 < Kk; k0 += 32) {
        if (k0 + 32 < Kk) {  // global_prefetch_b8 for next K-tile
            __builtin_prefetch(Ap + k0 + 32, 0, 1);
            __builtin_prefetch(Bp + k0 + 32, 0, 1);
        }
        v16bf af[4], bfv[4];
        #pragma unroll
        for (int i = 0; i < 4; ++i)
            af[i] = *reinterpret_cast<const v16bf*>(Ap + (size_t)i * 16 * Kk + k0);
        #pragma unroll
        for (int j = 0; j < 4; ++j)
            bfv[j] = *reinterpret_cast<const v16bf*>(Bp + (size_t)j * 16 * Kk + k0);
        #pragma unroll
        for (int i = 0; i < 4; ++i)
            #pragma unroll
            for (int j = 0; j < 4; ++j)
                acc[i][j] = wmma_bf16(af[i], bfv[j], acc[i][j]);
    }

    const int rowBase = m0 + (lane >> 4) * 8;
    const int colBase = n0 + lrow;
    if (BF16_OUT) {
        unsigned short* C = (unsigned short*)Cv;
        #pragma unroll
        for (int i = 0; i < 4; ++i)
            #pragma unroll
            for (int j = 0; j < 4; ++j)
                #pragma unroll
                for (int r = 0; r < 8; ++r)
                    C[(size_t)(rowBase + i * 16 + r) * Nn + colBase + j * 16] =
                        f2bf(acc[i][j][r]);
    } else {
        float* C = (float*)Cv;
        #pragma unroll
        for (int i = 0; i < 4; ++i)
            #pragma unroll
            for (int j = 0; j < 4; ++j)
                #pragma unroll
                for (int r = 0; r < 8; ++r)
                    C[(size_t)(rowBase + i * 16 + r) * Nn + colBase + j * 16] =
                        acc[i][j][r];
    }
}

// ---------- 5) Flash attention (causal, GQA) with TDM-staged K/V ----------
// 4 waves share one head's K/V stream: wave0 drives double-buffered TDM tile
// loads into padded LDS; all waves iterate to the block's causal limit (extra
// tiles are fully masked -> numerically inert).
__global__ __launch_bounds__(128) void k_attn(const unsigned short* __restrict__ Q,   // [M][DQ]
                                              const unsigned short* __restrict__ Kb,  // [M][DKV]
                                              const unsigned short* __restrict__ Vt,  // [B][DKV][S]
                                              unsigned short* __restrict__ Ab) {      // [M][DQ]
    __shared__ unsigned short kbuf[2][32 * KROW];    // K tile: 32 keys x 128 d (+pad)
    __shared__ unsigned short vbuf[2][128 * VROW];   // V tile: 128 d x 32 keys (+pad)
    __shared__ unsigned short pbuf[4][16 * 32];

    const int lane  = threadIdx.x & 31;
    const int wv    = threadIdx.x >> 5;
    const int b     = blockIdx.z;
    const int h     = blockIdx.y;
    const int kvh   = h >> 2;                        // Hq/Hkv = 4
    const int q0    = blockIdx.x * 64 + wv * 16;
    const int lrow  = lane & 15;
    const int lhalf = lane >> 4;
    const float kscale = 0.08838834764831845f * 1.4426950408889634f;  // 1/sqrt(D)*log2(e)

    // resident Q A-fragments
    v16bf qa[4];
    const unsigned short* Qp =
        Q + (size_t)(b * SEQ + q0 + lrow) * DQ + h * HEAD_DIM + lhalf * 16;
    #pragma unroll
    for (int di = 0; di < 4; ++di)
        qa[di] = *reinterpret_cast<const v16bf*>(Qp + di * 32);

    v8f o[8];
    #pragma unroll
    for (int j = 0; j < 8; ++j) o[j] = v8f{};
    float m_i[8], l_i[8];
    #pragma unroll
    for (int r = 0; r < 8; ++r) { m_i[r] = -INFINITY; l_i[r] = 0.f; }

    const unsigned short* Kg = Kb + (size_t)(b * SEQ) * DKV + kvh * HEAD_DIM;
    const unsigned short* Vg = Vt + ((size_t)b * DKV + kvh * HEAD_DIM) * SEQ;

    const int nt = blockIdx.x * 2 + 2;  // block-uniform tile count (32 keys each)

    if (wv == 0) {  // prime the pipeline: tile 0 -> buffers 0
        tdm_load_2d((uint32_t)(uintptr_t)&kbuf[0][0], Kg, 128, 32, DKV, 5, 7);
        tdm_load_2d((uint32_t)(uintptr_t)&vbuf[0][0], Vg, 32, 128, SEQ, 3, 7);
    }

    for (int t = 0; t < nt; ++t) {
        const int k0 = t * 32;
        const int cur = t & 1;
        if (wv == 0) {
            if (t + 1 < nt) {
                const int nxt = (t + 1) & 1;
                tdm_load_2d((uint32_t)(uintptr_t)&kbuf[nxt][0],
                            Kg + (size_t)(k0 + 32) * DKV, 128, 32, DKV, 5, 7);
                tdm_load_2d((uint32_t)(uintptr_t)&vbuf[nxt][0],
                            Vg + (size_t)(k0 + 32), 32, 128, SEQ, 3, 7);
                __builtin_amdgcn_s_wait_tensorcnt((short)2);  // tile t done
            } else {
                __builtin_amdgcn_s_wait_tensorcnt((short)0);
            }
        }
        __syncthreads();  // tile t visible to all waves

        // ---- preload all K fragments (one DS clause, pipelined waits) ----
        const unsigned short* kl = &kbuf[cur][0];
        v16bf kf[2][4];
        #pragma unroll
        for (int di = 0; di < 4; ++di) {
            kf[0][di] = *reinterpret_cast<const v16bf*>(
                kl + (size_t)lrow * KROW + di * 32 + lhalf * 16);
            kf[1][di] = *reinterpret_cast<const v16bf*>(
                kl + (size_t)(16 + lrow) * KROW + di * 32 + lhalf * 16);
        }
        // ---- S = Q K^T: two independent accumulator chains ----
        v8f s0 = v8f{}, s1 = v8f{};
        #pragma unroll
        for (int di = 0; di < 4; ++di) {
            s0 = wmma_bf16(qa[di], kf[0][di], s0);
            s1 = wmma_bf16(qa[di], kf[1][di], s1);
        }

        // ---- online softmax ----
        const int rowq = q0 + lhalf * 8;
        const int c0 = k0 + lrow;
        const int c1 = k0 + 16 + lrow;
        float p0[8], p1[8], alpha[8];
        #pragma unroll
        for (int r = 0; r < 8; ++r) {
            float v0 = (c0 <= rowq + r) ? s0[r] * kscale : -INFINITY;
            float v1 = (c1 <= rowq + r) ? s1[r] * kscale : -INFINITY;
            float mx = fmaxf(v0, v1);
            mx = fmaxf(mx, __shfl_xor(mx, 1));
            mx = fmaxf(mx, __shfl_xor(mx, 2));
            mx = fmaxf(mx, __shfl_xor(mx, 4));
            mx = fmaxf(mx, __shfl_xor(mx, 8));
            float mnew = fmaxf(m_i[r], mx);
            float a  = exp2f(m_i[r] - mnew);
            float e0 = exp2f(v0 - mnew);
            float e1 = exp2f(v1 - mnew);
            float sum = e0 + e1;
            sum += __shfl_xor(sum, 1);
            sum += __shfl_xor(sum, 2);
            sum += __shfl_xor(sum, 4);
            sum += __shfl_xor(sum, 8);
            l_i[r] = l_i[r] * a + sum;
            m_i[r] = mnew;
            alpha[r] = a;
            p0[r] = e0;
            p1[r] = e1;
        }
        #pragma unroll
        for (int j = 0; j < 8; ++j)
            #pragma unroll
            for (int r = 0; r < 8; ++r) o[j][r] *= alpha[r];

        // ---- relayout P (D-layout) -> A-fragment via per-wave LDS ----
        unsigned short* pb = pbuf[wv];
        #pragma unroll
        for (int r = 0; r < 8; ++r) {
            pb[(lhalf * 8 + r) * 32 + lrow]      = f2bf(p0[r]);
            pb[(lhalf * 8 + r) * 32 + 16 + lrow] = f2bf(p1[r]);
        }
        asm volatile("s_wait_dscnt 0" ::: "memory");
        v16bf pa = *reinterpret_cast<const v16bf*>(pb + lrow * 32 + lhalf * 16);

        // ---- preload all V fragments, then O += P V (8 independent chains) ----
        const unsigned short* vl = &vbuf[cur][0];
        v16bf vf[8];
        #pragma unroll
        for (int j = 0; j < 8; ++j)
            vf[j] = *reinterpret_cast<const v16bf*>(
                vl + (size_t)(j * 16 + lrow) * VROW + lhalf * 16);
        #pragma unroll
        for (int j = 0; j < 8; ++j)
            o[j] = wmma_bf16(pa, vf[j], o[j]);

        __syncthreads();  // all waves done reading tile t; buffers reusable
    }

    // ---- epilogue: normalize rows, store bf16 ----
    unsigned short* Op =
        Ab + (size_t)(b * SEQ + q0 + lhalf * 8) * DQ + h * HEAD_DIM;
    #pragma unroll
    for (int r = 0; r < 8; ++r) {
        float inv = 1.0f / l_i[r];
        #pragma unroll
        for (int j = 0; j < 8; ++j)
            Op[(size_t)r * DQ + j * 16 + lrow] = f2bf(o[j][r] * inv);
    }
}

// ---------- host launcher ----------
extern "C" void kernel_launch(void* const* d_in, const int* in_sizes, int n_in,
                              void* d_out, int out_size, void* d_ws, size_t ws_size,
                              hipStream_t stream) {
    const float* X  = (const float*)d_in[0];
    const float* Wq = (const float*)d_in[1];
    const float* Wk = (const float*)d_in[2];
    const float* Wv = (const float*)d_in[3];
    const float* Wo = (const float*)d_in[4];

    char* p = (char*)d_ws;
    unsigned short* Xb  = (unsigned short*)p; p += (size_t)MROWS * HID * 2;
    unsigned short* WqT = (unsigned short*)p; p += (size_t)DQ * HID * 2;
    unsigned short* WkT = (unsigned short*)p; p += (size_t)DKV * HID * 2;
    unsigned short* WvT = (unsigned short*)p; p += (size_t)DKV * HID * 2;
    unsigned short* WoT = (unsigned short*)p; p += (size_t)HID * DQ * 2;
    unsigned short* Qb  = (unsigned short*)p; p += (size_t)MROWS * DQ * 2;
    unsigned short* Kb  = (unsigned short*)p; p += (size_t)MROWS * DKV * 2;
    unsigned short* Vb  = (unsigned short*)p; p += (size_t)MROWS * DKV * 2;
    unsigned short* Vt  = (unsigned short*)p; p += (size_t)BATCH * DKV * SEQ * 2;
    unsigned short* Abf = (unsigned short*)p; p += (size_t)MROWS * DQ * 2;

    float* out = (float*)d_out;
    const dim3 tb(32, 8);

    {
        int n = MROWS * HID;
        k_f32_to_bf16<<<n / 8 / 256, 256, 0, stream>>>(X, Xb, n);
    }
    k_transpose_w<<<dim3(DQ / 32,  HID / 32), tb, 0, stream>>>(Wq, WqT, HID, DQ);
    k_transpose_w<<<dim3(DKV / 32, HID / 32), tb, 0, stream>>>(Wk, WkT, HID, DKV);
    k_transpose_w<<<dim3(DKV / 32, HID / 32), tb, 0, stream>>>(Wv, WvT, HID, DKV);
    k_transpose_w<<<dim3(HID / 32, DQ / 32),  tb, 0, stream>>>(Wo, WoT, DQ, HID);

    k_gemm<true><<<dim3(DQ / 128,  MROWS / 128), 128, 0, stream>>>(Xb, WqT, Qb, MROWS, DQ,  HID);
    k_gemm<true><<<dim3(DKV / 128, MROWS / 128), 128, 0, stream>>>(Xb, WkT, Kb, MROWS, DKV, HID);
    k_gemm<true><<<dim3(DKV / 128, MROWS / 128), 128, 0, stream>>>(Xb, WvT, Vb, MROWS, DKV, HID);

    k_transpose_v<<<dim3(DKV / 32, SEQ / 32, BATCH), tb, 0, stream>>>(Vb, Vt);

    k_attn<<<dim3(SEQ / 64, NUM_HEADS, BATCH), 128, 0, stream>>>(Qb, Kb, Vt, Abf);

    k_gemm<false><<<dim3(HID / 128, MROWS / 128), 128, 0, stream>>>(Abf, WoT, out, MROWS, HID, DQ);
}
